// ISDT_19095424598412
// MI455X (gfx1250) — compile-verified, matrix operand order checked by reading, not run on hardware
//
#include <hip/hip_runtime.h>
#include <hip/hip_bf16.h>
#include <stdint.h>

// ---------------------------------------------------------------------------
// CDNA5 / gfx1250 implementation: bf16 WMMA GEMM pipeline for VQ encoder.
//  - v_wmma_f32_16x16x32_bf16 for all GEMMs (f32 A converted while staging)
//  - B tiles staged with global_load_async_to_lds_b128 (ASYNCcnt path)
//  - fused similarity-GEMM + row argmax (sim matrix never hits memory)
// ---------------------------------------------------------------------------

typedef __bf16 bf16_t;
typedef __attribute__((ext_vector_type(16))) __bf16 v16bf;
typedef __attribute__((ext_vector_type(8)))  float  v8f;

union FragBF { v16bf v; uint4 q[2]; };
union Pack4  { bf16_t b[4]; uint2 u; };

static __device__ __forceinline__ float wave_sum32(float s) {
#pragma unroll
  for (int off = 16; off > 0; off >>= 1) s += __shfl_xor(s, off, 32);
  return s;
}

static __device__ __forceinline__ uint32_t lds_addr(const void* p) {
  // generic -> LDS addrspace is a low-32-bit truncation on amdgcn
  return (uint32_t)(uintptr_t)p;
}

static __device__ __forceinline__ void async_b128_to_lds(uint32_t lds_off,
                                                         const void* gptr) {
  asm volatile("global_load_async_to_lds_b128 %0, %1, off"
               :: "v"(lds_off), "v"((unsigned long long)(uintptr_t)gptr)
               : "memory");
}

static __device__ __forceinline__ void wait_async0() {
  asm volatile("s_wait_asynccnt 0x0" ::: "memory");
}

static __device__ __forceinline__ void cvt_store4(bf16_t* dst, float4 f) {
  Pack4 pk;
  pk.b[0] = (bf16_t)f.x; pk.b[1] = (bf16_t)f.y;
  pk.b[2] = (bf16_t)f.z; pk.b[3] = (bf16_t)f.w;
  *(uint2*)dst = pk.u;
}

// --------------------------- weight transpose+cvt ---------------------------
// W [K][N] f32 (row-major)  ->  Wt [N][K] bf16 (row-major)  => NT GEMM layout
__global__ void k_transpose_cvt(const float* __restrict__ W, bf16_t* __restrict__ Wt,
                                int K, int N) {
  int i = blockIdx.x * 256 + threadIdx.x;
  if (i < K * N) {
    int k = i / N, n = i - k * N;
    Wt[(size_t)n * K + k] = (bf16_t)W[i];
  }
}

// ------------------------------- WMMA GEMM NT -------------------------------
// C[M][N] = act(A[M][K] @ Bt[N][K]^T + bias), A f32 (cvt->bf16 in LDS), Bt bf16.
// Block: 256 threads (8 waves), tile 128(M) x 128(N), K-step 32.
// Waves: wave_m = wid&3 (4 x 32 rows), wave_n = wid>>2 (2 x 64 cols).
__global__ void __launch_bounds__(256, 1)
k_gemm_nt(const float* __restrict__ A, const bf16_t* __restrict__ Bt,
          const float* __restrict__ bias, float* __restrict__ C,
          int M, int N, int K, int relu) {
  __shared__ __align__(16) bf16_t sA[128 * 32];
  __shared__ __align__(16) bf16_t sB[128 * 32];

  const int t = threadIdx.x;
  const int wid = t >> 5, lane = t & 31, half = lane >> 4, l15 = lane & 15;
  const int wave_m = wid & 3, wave_n = wid >> 2;
  const int m0 = blockIdx.y * 128, n0 = blockIdx.x * 128;

  // staging geometry (thread-constant)
  const int arow = t >> 3, acq = t & 7;     // A: rows arow+32i, float4 slot acq
  const int brow = t >> 2, bq  = t & 3;     // B: rows brow+64i, uint4 slot bq
  const float*  Abase = A  + (size_t)(m0 + arow) * K + acq * 4;
  const bf16_t* Bbase = Bt + (size_t)(n0 + brow) * K + bq * 8;
  bf16_t* sA0 = &sA[arow * 32 + acq * 4];
  const uint32_t ldsB0 = lds_addr(&sB[(brow     ) * 32 + bq * 8]);
  const uint32_t ldsB1 = lds_addr(&sB[(brow + 64) * 32 + bq * 8]);

  const v8f vzero = {0.f, 0.f, 0.f, 0.f, 0.f, 0.f, 0.f, 0.f};
  v8f acc[2][4];
#pragma unroll
  for (int mi = 0; mi < 2; ++mi)
#pragma unroll
    for (int ni = 0; ni < 4; ++ni) acc[mi][ni] = vzero;

  for (int k0 = 0; k0 < K; k0 += 32) {
    // ---- B tile [128][32] bf16: async copy global -> LDS (no VGPR transit) ----
    async_b128_to_lds(ldsB0, Bbase + k0);
    async_b128_to_lds(ldsB1, Bbase + (size_t)64 * K + k0);

    // ---- A tile [128][32]: batched f32 loads -> cvt bf16 -> LDS ----
    float4 f0 = *(const float4*)(Abase + k0);
    float4 f1 = *(const float4*)(Abase + (size_t)32 * K + k0);
    float4 f2 = *(const float4*)(Abase + (size_t)64 * K + k0);
    float4 f3 = *(const float4*)(Abase + (size_t)96 * K + k0);
    __builtin_prefetch(Abase + k0 + 32, 0, 1);                   // speculative
    __builtin_prefetch(Abase + (size_t)64 * K + k0 + 32, 0, 1);
    cvt_store4(sA0,             f0);
    cvt_store4(sA0 + 32 * 32,   f1);
    cvt_store4(sA0 + 64 * 32,   f2);
    cvt_store4(sA0 + 96 * 32,   f3);

    wait_async0();
    __syncthreads();

    // ---- fragments (ISA 16-bit 16x32 lane layout) + 8 WMMAs ----
    FragBF fa[2], fb[4];
#pragma unroll
    for (int mi = 0; mi < 2; ++mi) {
      int ar = wave_m * 32 + mi * 16 + l15;
      fa[mi].q[0] = *(const uint4*)&sA[ar * 32 + half * 8];
      fa[mi].q[1] = *(const uint4*)&sA[ar * 32 + 16 + half * 8];
    }
#pragma unroll
    for (int ni = 0; ni < 4; ++ni) {
      int br = wave_n * 64 + ni * 16 + l15;
      fb[ni].q[0] = *(const uint4*)&sB[br * 32 + half * 8];
      fb[ni].q[1] = *(const uint4*)&sB[br * 32 + 16 + half * 8];
    }
#pragma unroll
    for (int mi = 0; mi < 2; ++mi)
#pragma unroll
      for (int ni = 0; ni < 4; ++ni)
        acc[mi][ni] = __builtin_amdgcn_wmma_f32_16x16x32_bf16(
            false, fa[mi].v, false, fb[ni].v, (short)0, acc[mi][ni], false, false);
    __syncthreads();
  }

  // ---- epilogue: bias + relu, f32 out ----
#pragma unroll
  for (int mi = 0; mi < 2; ++mi)
#pragma unroll
    for (int ni = 0; ni < 4; ++ni)
#pragma unroll
      for (int r = 0; r < 8; ++r) {
        int row = m0 + wave_m * 32 + mi * 16 + half * 8 + r;
        int col = n0 + wave_n * 64 + ni * 16 + l15;
        float v = acc[mi][ni][r] + bias[col];
        if (relu) v = fmaxf(v, 0.f);
        C[(size_t)row * N + col] = v;
      }
}

// ------------------------- fused sim GEMM + argmax --------------------------
// sim = Xn[32768][512] @ En[1024][512]^T ; codes = argmax_n sim (== argmin dist)
// Block: 8 waves, 128 rows (16 rows/wave), loop 8 panels of 128 codes.
__global__ void __launch_bounds__(256, 1)
k_sim_argmax(const float* __restrict__ Xn, const bf16_t* __restrict__ En,
             float* __restrict__ codes, int* __restrict__ idx_ws, int head) {
  __shared__ __align__(16) bf16_t sA[128 * 32];
  __shared__ __align__(16) bf16_t sB[128 * 32];

  const int t = threadIdx.x;
  const int wid = t >> 5, lane = t & 31, half = lane >> 4, l15 = lane & 15;
  const int m0 = blockIdx.x * 128;
  const int D = 512;

  const int arow = t >> 3, acq = t & 7;
  const int brow = t >> 2, bq  = t & 3;
  const float* Abase = Xn + (size_t)(m0 + arow) * D + acq * 4;
  bf16_t* sA0 = &sA[arow * 32 + acq * 4];
  const uint32_t ldsB0 = lds_addr(&sB[(brow     ) * 32 + bq * 8]);
  const uint32_t ldsB1 = lds_addr(&sB[(brow + 64) * 32 + bq * 8]);

  float rmax[8]; int ridx[8];
#pragma unroll
  for (int r = 0; r < 8; ++r) { rmax[r] = -3.4e38f; ridx[r] = 0; }
  const v8f vzero = {0.f, 0.f, 0.f, 0.f, 0.f, 0.f, 0.f, 0.f};

  for (int p = 0; p < 8; ++p) {          // 8 panels x 128 codes = 1024 codes
    const bf16_t* Bbase = En + (size_t)(p * 128 + brow) * D + bq * 8;
    v8f acc[8];
#pragma unroll
    for (int ni = 0; ni < 8; ++ni) acc[ni] = vzero;

    for (int k0 = 0; k0 < D; k0 += 32) {
      async_b128_to_lds(ldsB0, Bbase + k0);
      async_b128_to_lds(ldsB1, Bbase + (size_t)64 * D + k0);

      float4 f0 = *(const float4*)(Abase + k0);
      float4 f1 = *(const float4*)(Abase + (size_t)32 * D + k0);
      float4 f2 = *(const float4*)(Abase + (size_t)64 * D + k0);
      float4 f3 = *(const float4*)(Abase + (size_t)96 * D + k0);
      cvt_store4(sA0,           f0);
      cvt_store4(sA0 + 32 * 32, f1);
      cvt_store4(sA0 + 64 * 32, f2);
      cvt_store4(sA0 + 96 * 32, f3);

      wait_async0();
      __syncthreads();

      FragBF fa;
      int ar = wid * 16 + l15;
      fa.q[0] = *(const uint4*)&sA[ar * 32 + half * 8];
      fa.q[1] = *(const uint4*)&sA[ar * 32 + 16 + half * 8];
#pragma unroll
      for (int ni = 0; ni < 8; ++ni) {
        FragBF fb;
        int br = ni * 16 + l15;
        fb.q[0] = *(const uint4*)&sB[br * 32 + half * 8];
        fb.q[1] = *(const uint4*)&sB[br * 32 + 16 + half * 8];
        acc[ni] = __builtin_amdgcn_wmma_f32_16x16x32_bf16(
            false, fa.v, false, fb.v, (short)0, acc[ni], false, false);
      }
      __syncthreads();
    }

    // ---- per-row argmax update for this 128-code panel ----
#pragma unroll
    for (int r = 0; r < 8; ++r) {
      float best = acc[0][r];
      int bi = p * 128 + l15;
#pragma unroll
      for (int ni = 1; ni < 8; ++ni) {
        float v = acc[ni][r];
        int vi = p * 128 + ni * 16 + l15;
        if (v > best) { best = v; bi = vi; }
      }
#pragma unroll
      for (int off = 1; off < 16; off <<= 1) {   // butterfly within 16-lane half
        float ov = __shfl_xor(best, off, 32);
        int   oi = __shfl_xor(bi, off, 32);
        if (ov > best || (ov == best && oi < bi)) { best = ov; bi = oi; }
      }
      if (best > rmax[r]) { rmax[r] = best; ridx[r] = bi; }
    }
  }

  if (l15 == 0) {
#pragma unroll
    for (int r = 0; r < 8; ++r) {
      int row = m0 + wid * 16 + half * 8 + r;
      codes[(size_t)row * 3 + head] = (float)ridx[r];
      idx_ws[row] = ridx[r];
    }
  }
}

// -------------------------- row L2-norm (D = 512) ---------------------------
__global__ void k_rownorm(const float* __restrict__ Z, float* __restrict__ Xn) {
  const int wid = threadIdx.x >> 5, lane = threadIdx.x & 31;
  const int row = blockIdx.x * 8 + wid;
  const float* z = Z + (size_t)row * 512;
  float v[16], s = 0.f;
#pragma unroll
  for (int i = 0; i < 16; ++i) { v[i] = z[lane + 32 * i]; s += v[i] * v[i]; }
  s = wave_sum32(s);
  float sc = rsqrtf(s + 1e-12f);
  float* x = Xn + (size_t)row * 512;
#pragma unroll
  for (int i = 0; i < 16; ++i) x[lane + 32 * i] = v[i] * sc;
}

// --------------------- codebook normalize: f32 + bf16 copies ----------------
__global__ void k_cbnorm(const float* __restrict__ CB, float* __restrict__ Ef,
                         bf16_t* __restrict__ Eb) {
  const int wid = threadIdx.x >> 5, lane = threadIdx.x & 31;
  const int row = blockIdx.x * 8 + wid;
  const float* z = CB + (size_t)row * 512;
  float v[16], s = 0.f;
#pragma unroll
  for (int i = 0; i < 16; ++i) { v[i] = z[lane + 32 * i]; s += v[i] * v[i]; }
  s = wave_sum32(s);
  float sc = rsqrtf(s + 1e-12f);
#pragma unroll
  for (int i = 0; i < 16; ++i) {
    float q = v[i] * sc;
    Ef[(size_t)row * 512 + lane + 32 * i] = q;
    Eb[(size_t)row * 512 + lane + 32 * i] = (bf16_t)q;
  }
}

// -------------------- alpha = sigmoid(H @ Wk + bk), per-row ------------------
__global__ void k_alpha(const float* __restrict__ H, const float* __restrict__ Wk,
                        const float* __restrict__ bk, float* __restrict__ alpha) {
  const int wid = threadIdx.x >> 5, lane = threadIdx.x & 31;
  const int row = blockIdx.x * 8 + wid;
  const float* h = H + (size_t)row * 512;
  float s = 0.f;
#pragma unroll
  for (int i = 0; i < 16; ++i) s += h[lane + 32 * i] * Wk[lane + 32 * i];
  s = wave_sum32(s);
  if (lane == 0) alpha[row] = 1.f / (1.f + __expf(-(s + bk[0])));
}

// ---------------------- top-32 (single block, iterative) ---------------------
__global__ void k_topk(const float* __restrict__ alpha, float* __restrict__ tmp,
                       float* __restrict__ keyidx) {
  __shared__ float sm[256];
  __shared__ int   si[256];
  const int t = threadIdx.x;
  for (int j = t; j < 32768; j += 256) tmp[j] = alpha[j];
  __threadfence_block();
  __syncthreads();
  for (int round = 0; round < 32; ++round) {
    float best = -3.4e38f; int bi = 0;
    for (int j = t; j < 32768; j += 256) {
      float v = tmp[j];
      if (v > best) { best = v; bi = j; }
    }
    sm[t] = best; si[t] = bi;
    __syncthreads();
    for (int s = 128; s > 0; s >>= 1) {
      if (t < s) {
        if (sm[t + s] > sm[t] || (sm[t + s] == sm[t] && si[t + s] < si[t])) {
          sm[t] = sm[t + s]; si[t] = si[t + s];
        }
      }
      __syncthreads();
    }
    if (t == 0) { keyidx[round] = (float)si[0]; tmp[si[0]] = -3.4e38f; }
    __threadfence_block();
    __syncthreads();
  }
}

// --------------- gather q = en[idx] -> token_emb + commit loss ---------------
__global__ void k_gather_commit(const int* __restrict__ idx, const float* __restrict__ Ef,
                                const float* __restrict__ Xn, float* __restrict__ tok,
                                float* __restrict__ sums, int head) {
  const int wid = threadIdx.x >> 5, lane = threadIdx.x & 31;
  const int row = blockIdx.x * 8 + wid;
  const int id = idx[row];
  const float* q = Ef + (size_t)id * 512;
  const float* x = Xn + (size_t)row * 512;
  float* o = tok + (size_t)row * 1536 + head * 512;
  float s = 0.f;
#pragma unroll
  for (int i = 0; i < 16; ++i) {
    int c = lane + 32 * i;
    float qv = q[c];
    o[c] = qv;
    float d = qv - x[c];
    s += d * d;
  }
  s = wave_sum32(s);
  if (lane == 0) atomicAdd(&sums[head], s);
}

__global__ void k_init_sums(float* sums) {
  if (threadIdx.x < 3) sums[threadIdx.x] = 0.f;
}

__global__ void k_finalize(const float* __restrict__ sums, float* __restrict__ lvq) {
  lvq[0] = 0.25f * (sums[0] + sums[1] + sums[2]) * (1.f / (32768.f * 512.f));
}

// ------------------------------- host launcher -------------------------------
extern "C" void kernel_launch(void* const* d_in, const int* in_sizes, int n_in,
                              void* d_out, int out_size, void* d_ws, size_t ws_size,
                              hipStream_t stream) {
  (void)in_sizes; (void)n_in; (void)out_size; (void)ws_size;
  const float* h0 = (const float*)d_in[0];
  const float* W1 = (const float*)d_in[1];  const float* b1 = (const float*)d_in[2];
  const float* W2 = (const float*)d_in[3];  const float* b2 = (const float*)d_in[4];
  const float* Wm = (const float*)d_in[5];  const float* bm = (const float*)d_in[6];
  const float* Wt = (const float*)d_in[7];  const float* bt = (const float*)d_in[8];
  const float* Wp = (const float*)d_in[9];  const float* bp = (const float*)d_in[10];
  const float* cbm = (const float*)d_in[11];
  const float* cbt = (const float*)d_in[12];
  const float* cbp = (const float*)d_in[13];
  const float* Wk = (const float*)d_in[14]; const float* bk = (const float*)d_in[15];

  float* out = (float*)d_out;
  const size_t OFF_CODES = 16777216;   // after H [32768*512]
  const size_t OFF_KEY   = 16875520;   // after codes [32768*3]
  const size_t OFF_TOKEN = 16875552;   // after key_idx [32]
  const size_t OFF_LVQ   = 67207200;   // after token_emb [32768*1536]

  char* ws = (char*)d_ws;
  size_t o = 0;
  bf16_t* W1t = (bf16_t*)(ws + o); o += (size_t)512 * 1024 * 2;
  bf16_t* W2t = (bf16_t*)(ws + o); o += (size_t)512 * 512 * 2;
  bf16_t* Wmt = (bf16_t*)(ws + o); o += (size_t)512 * 512 * 2;
  bf16_t* Wtt = (bf16_t*)(ws + o); o += (size_t)512 * 512 * 2;
  bf16_t* Wpt = (bf16_t*)(ws + o); o += (size_t)512 * 512 * 2;
  o = (o + 255) & ~(size_t)255;
  float* A1  = (float*)(ws + o); o += (size_t)32768 * 512 * 4;
  float* Z   = (float*)(ws + o); o += (size_t)32768 * 512 * 4;
  float* Xn  = (float*)(ws + o); o += (size_t)32768 * 512 * 4;
  float* Ef  = (float*)(ws + o); o += (size_t)1024 * 512 * 4;
  bf16_t* Eb = (bf16_t*)(ws + o); o += (size_t)1024 * 512 * 2;
  float* alpha  = (float*)(ws + o); o += (size_t)32768 * 4;
  float* atmp   = (float*)(ws + o); o += (size_t)32768 * 4;
  int*   idxw   = (int*)(ws + o);   o += (size_t)32768 * 4;
  float* sums   = (float*)(ws + o); o += 16;

  k_init_sums<<<1, 32, 0, stream>>>(sums);

  // weight transposes (f32 -> bf16, [K][N] -> [N][K])
  k_transpose_cvt<<<(1024 * 512) / 256, 256, 0, stream>>>(W1, W1t, 1024, 512);
  k_transpose_cvt<<<(512 * 512) / 256, 256, 0, stream>>>(W2, W2t, 512, 512);
  k_transpose_cvt<<<(512 * 512) / 256, 256, 0, stream>>>(Wm, Wmt, 512, 512);
  k_transpose_cvt<<<(512 * 512) / 256, 256, 0, stream>>>(Wt, Wtt, 512, 512);
  k_transpose_cvt<<<(512 * 512) / 256, 256, 0, stream>>>(Wp, Wpt, 512, 512);

  // encoder
  k_gemm_nt<<<dim3(4, 256), 256, 0, stream>>>(h0, W1t, b1, A1, 32768, 512, 1024, 1);
  k_gemm_nt<<<dim3(4, 256), 256, 0, stream>>>(A1, W2t, b2, out, 32768, 512, 512, 1); // H -> d_out

  // alpha + top-32
  k_alpha<<<4096, 256, 0, stream>>>(out, Wk, bk, alpha);
  k_topk<<<1, 256, 0, stream>>>(alpha, atmp, out + OFF_KEY);

  // three VQ heads (buffers reused sequentially)
  const float* cbs[3]   = {cbm, cbt, cbp};
  const bf16_t* wts[3]  = {Wmt, Wtt, Wpt};
  const float* bias3[3] = {bm, bt, bp};
  for (int h = 0; h < 3; ++h) {
    k_cbnorm<<<128, 256, 0, stream>>>(cbs[h], Ef, Eb);
    k_gemm_nt<<<dim3(4, 256), 256, 0, stream>>>(out, wts[h], bias3[h], Z,
                                                32768, 512, 512, 0);
    k_rownorm<<<4096, 256, 0, stream>>>(Z, Xn);
    k_sim_argmax<<<256, 256, 0, stream>>>(Xn, Eb, out + OFF_CODES, idxw, h);
    k_gather_commit<<<4096, 256, 0, stream>>>(idxw, Ef, Xn, out + OFF_TOKEN, sums, h);
  }

  k_finalize<<<1, 1, 0, stream>>>(sums, out + OFF_LVQ);
}
